// BatchGraphTripleConv_22110491640377
// MI455X (gfx1250) — compile-verified, matrix operand order checked by reading, not run on hardware
//
#include <hip/hip_runtime.h>
#include <hip/hip_bf16.h>

// ---------------------------------------------------------------------------
// Problem constants (reference: B=64, N=256, T=512, DO=DP=H=DOUT=512)
// ---------------------------------------------------------------------------
constexpr int kB    = 64;
constexpr int kN    = 256;
constexpr int kT    = 512;
constexpr int kDO   = 512;
constexpr int kH    = 512;
constexpr int kDOUT = 512;
constexpr int kM1   = kB * kT;   // 32768 edge rows
constexpr int kM2   = kB * kN;   // 16384 object rows
constexpr int kK1   = 2 * kDO + kH;      // 1536
constexpr int kN2   = 2 * kH + kDOUT;    // 1536

typedef __attribute__((ext_vector_type(16))) __bf16 v16bf;
typedef __attribute__((ext_vector_type(8)))  float  v8f;
typedef __attribute__((ext_vector_type(4)))  int    v4i;

#if __has_builtin(__builtin_amdgcn_global_load_async_to_lds_b128) && \
    __has_builtin(__builtin_amdgcn_s_wait_asynccnt)
#define ASYNC_COPY 1
#else
#define ASYNC_COPY 0
#endif

#define AS1 __attribute__((address_space(1)))
#define AS3 __attribute__((address_space(3)))

__device__ __forceinline__ unsigned short f2bf(float f) {
    unsigned int u = __float_as_uint(f);
    // round-to-nearest-even bf16
    u = (u + 0x7FFFu + ((u >> 16) & 1u)) >> 16;
    return (unsigned short)u;
}

__device__ __forceinline__ v16bf ld_frag(const unsigned short* p) {
    union { v16bf v; uint4 q[2]; } u;
    u.q[0] = *(const uint4*)(p);
    u.q[1] = *(const uint4*)(p + 8);
    return u.v;
}

#if ASYNC_COPY
__device__ __forceinline__ void async_ld_b128(const unsigned short* g, unsigned short* l) {
    __builtin_amdgcn_global_load_async_to_lds_b128(
        (AS1 v4i*)g, (AS3 v4i*)l, 0, 0);
}
#endif

// ---------------------------------------------------------------------------
// Weight convert: w (K x N, f32, row-major)  ->  wt (N x K, bf16, row-major)
// ---------------------------------------------------------------------------
__global__ void k_w_cvt(const float* __restrict__ w, unsigned short* __restrict__ wt,
                        int K, int N) {
    int i = blockIdx.x * 256 + threadIdx.x;     // linear over output N*K
    if (i >= K * N) return;
    int n = i / K;
    int k = i - n * K;
    wt[i] = f2bf(w[(long long)k * N + n]);
}

// ---------------------------------------------------------------------------
// Gather + f32->bf16: cur_t[m] = [obj[b,s], pred[b,t], obj[b,o]]   (m = b*T+t)
// ---------------------------------------------------------------------------
__global__ void k_gather(const float* __restrict__ obj, const float* __restrict__ pred,
                         const long long* __restrict__ edges,
                         unsigned short* __restrict__ curT) {
    int m = blockIdx.x;                 // 0..B*T-1
    int b = m >> 9;                     // T = 512
    long long s = edges[(long long)m * 3 + 0];
    long long o = edges[(long long)m * 3 + 2];
    const float* ps = obj  + ((long long)b * kN + s) * kDO;
    const float* pp = pred + (long long)m * kDO;
    const float* po = obj  + ((long long)b * kN + o) * kDO;
    unsigned short* out = curT + (long long)m * kK1;
    for (int c = threadIdx.x; c < kDO; c += 256) {
        out[c]           = f2bf(ps[c]);
        out[kDO + c]     = f2bf(pp[c]);
        out[2 * kDO + c] = f2bf(po[c]);
    }
}

__global__ void k_zero(float* __restrict__ p, long long n) {
    long long i = (long long)blockIdx.x * 256 + threadIdx.x;
    if (i < n) p[i] = 0.0f;
}

__global__ void k_counts(const long long* __restrict__ edges, float* __restrict__ counts) {
    int m = blockIdx.x * 256 + threadIdx.x;
    if (m >= kM1) return;
    int b = m >> 9;
    int s = (int)edges[(long long)m * 3 + 0];
    int o = (int)edges[(long long)m * 3 + 2];
    atomicAdd(&counts[b * kN + s], 1.0f);
    atomicAdd(&counts[b * kN + o], 1.0f);
}

// pooled /= max(counts,1), convert to bf16 for GEMM3 input
__global__ void k_norm(const float* __restrict__ pooled, const float* __restrict__ counts,
                       unsigned short* __restrict__ pbf) {
    long long i = (long long)blockIdx.x * 256 + threadIdx.x;
    if (i >= (long long)kM2 * kH) return;
    int objrow = (int)(i >> 9);         // H = 512
    float c = fmaxf(counts[objrow], 1.0f);
    pbf[i] = f2bf(pooled[i] / c);
}

// ---------------------------------------------------------------------------
// WMMA GEMM:  C = relu(A(bf16, MxK) * Bt(bf16, NxK)^T + bias)
//   128x128 block tile, BK=32, 256 threads (8 waves, 2x4), wave tile 64x32.
//   Double-buffered LDS; async global->LDS when the toolchain exposes it.
//   EPI 0: write f32   EPI 1: write bf16   EPI 2: scatter (GEMM2 epilogue)
// ---------------------------------------------------------------------------
enum { EPI_F32 = 0, EPI_BF16 = 1, EPI_SCATTER = 2 };

template <int EPI>
__global__ __launch_bounds__(256)
void k_gemm(const unsigned short* __restrict__ A,
            const unsigned short* __restrict__ Bt,
            const float* __restrict__ bias,
            int Ncols, int K,
            float* __restrict__ outF,
            unsigned short* __restrict__ outH,
            const long long* __restrict__ edges,
            float* __restrict__ pooled,
            float* __restrict__ outP) {
    constexpr int BM = 128, BN = 128, BK = 32, PITCH = BK + 8;   // +8 bf16 pad
    __shared__ unsigned short As[2][BM * PITCH];
    __shared__ unsigned short Bs[2][BN * PITCH];
    __shared__ int sS[BM];
    __shared__ int sO[BM];

    const int tid  = threadIdx.x;
    const int row0 = blockIdx.x * BM;
    const int col0 = blockIdx.y * BN;

    if (EPI == EPI_SCATTER) {
        if (tid < BM) {
            int m = row0 + tid;
            sS[tid] = (int)edges[(long long)m * 3 + 0];
            sO[tid] = (int)edges[(long long)m * 3 + 2];
        }
    }

    const int lane  = tid & 31;
    const int w     = tid >> 5;
    const int wr    = w >> 2;           // 0..1  -> 64 rows
    const int wc    = w & 3;            // 0..3  -> 32 cols
    const int lhalf = lane >> 4;        // K/M half select
    const int l15   = lane & 15;

    // per-thread tile-copy coordinates: 512 b128 per operand tile, 2 per thread
    const int cr = tid >> 2;            // 0..63  (row, +64 for second pass)
    const int cc = (tid & 3) * 8;       // ushort column offset of the uint4

    v8f zero = {0.f, 0.f, 0.f, 0.f, 0.f, 0.f, 0.f, 0.f};
    v8f acc[4][2];
#pragma unroll
    for (int i = 0; i < 4; ++i)
#pragma unroll
        for (int j = 0; j < 2; ++j) acc[i][j] = zero;

    const int nK = K / BK;

#if ASYNC_COPY
    // ---- async global->LDS double-buffered pipeline -----------------------
    auto aload = [&](int buf, int kk) {
#pragma unroll
        for (int p = 0; p < 2; ++p) {
            int r = cr + p * 64;
            async_ld_b128(A  + (long long)(row0 + r) * K + kk + cc, &As[buf][r * PITCH + cc]);
            async_ld_b128(Bt + (long long)(col0 + r) * K + kk + cc, &Bs[buf][r * PITCH + cc]);
        }
    };
    aload(0, 0);
    for (int kt = 0; kt < nK; ++kt) {
        __builtin_amdgcn_s_wait_asynccnt(0);   // current tile landed in LDS
        __syncthreads();                       // visible to all waves; prev reads done
        if (kt + 1 < nK) aload((kt + 1) & 1, (kt + 1) * BK);

        const unsigned short* as = As[kt & 1];
        const unsigned short* bs = Bs[kt & 1];
        v16bf af[4], bfr[2];
#pragma unroll
        for (int i = 0; i < 4; ++i)
            af[i] = ld_frag(as + (wr * 64 + i * 16 + l15) * PITCH + lhalf * 16);
#pragma unroll
        for (int j = 0; j < 2; ++j)
            bfr[j] = ld_frag(bs + (wc * 32 + j * 16 + l15) * PITCH + lhalf * 16);
#pragma unroll
        for (int i = 0; i < 4; ++i)
#pragma unroll
            for (int j = 0; j < 2; ++j)
                acc[i][j] = __builtin_amdgcn_wmma_f32_16x16x32_bf16(
                    false, af[i], false, bfr[j], (short)0, acc[i][j], false, false);
    }
#else
    // ---- register-prefetch double-buffered pipeline -----------------------
    uint4 ra[2], rb[2];
    auto gload = [&](int kk) {
#pragma unroll
        for (int p = 0; p < 2; ++p) {
            int r = cr + p * 64;
            ra[p] = *(const uint4*)(A  + (long long)(row0 + r) * K + kk + cc);
            rb[p] = *(const uint4*)(Bt + (long long)(col0 + r) * K + kk + cc);
        }
    };
    auto sstore = [&](int buf) {
#pragma unroll
        for (int p = 0; p < 2; ++p) {
            int r = cr + p * 64;
            *(uint4*)(&As[buf][r * PITCH + cc]) = ra[p];
            *(uint4*)(&Bs[buf][r * PITCH + cc]) = rb[p];
        }
    };
    gload(0);
    sstore(0);
    for (int kt = 0; kt < nK; ++kt) {
        __syncthreads();                       // buf[kt&1] ready; old reads done
        if (kt + 1 < nK) gload((kt + 1) * BK); // overlap global latency with WMMA

        const unsigned short* as = As[kt & 1];
        const unsigned short* bs = Bs[kt & 1];
        v16bf af[4], bfr[2];
#pragma unroll
        for (int i = 0; i < 4; ++i)
            af[i] = ld_frag(as + (wr * 64 + i * 16 + l15) * PITCH + lhalf * 16);
#pragma unroll
        for (int j = 0; j < 2; ++j)
            bfr[j] = ld_frag(bs + (wc * 32 + j * 16 + l15) * PITCH + lhalf * 16);
#pragma unroll
        for (int i = 0; i < 4; ++i)
#pragma unroll
            for (int j = 0; j < 2; ++j)
                acc[i][j] = __builtin_amdgcn_wmma_f32_16x16x32_bf16(
                    false, af[i], false, bfr[j], (short)0, acc[i][j], false, false);

        if (kt + 1 < nK) sstore((kt + 1) & 1); // target buffer last read at kt-1
    }
#endif

    // epilogue: C layout — VGPR r: M = r + 8*(lane>=16), N = lane&15
#pragma unroll
    for (int j = 0; j < 2; ++j) {
        const int n = col0 + wc * 32 + j * 16 + l15;
        const float bj = bias[n];
#pragma unroll
        for (int i = 0; i < 4; ++i) {
#pragma unroll
            for (int r = 0; r < 8; ++r) {
                int lr = wr * 64 + i * 16 + lhalf * 8 + r;   // local row
                int m  = row0 + lr;
                float v = fmaxf(acc[i][j][r] + bj, 0.0f);
                if (EPI == EPI_F32) {
                    outF[(long long)m * Ncols + n] = v;
                } else if (EPI == EPI_BF16) {
                    outH[(long long)m * Ncols + n] = f2bf(v);
                } else {
                    int b = m >> 9;                          // T = 512
                    if (n < kH) {
                        atomicAdd(&pooled[((long long)b * kN + sS[lr]) * kH + n], v);
                    } else if (n < kH + kDOUT) {
                        outP[(long long)m * kDOUT + (n - kH)] = v;
                    } else {
                        atomicAdd(&pooled[((long long)b * kN + sO[lr]) * kH + (n - kH - kDOUT)], v);
                    }
                }
            }
        }
    }
}

// ---------------------------------------------------------------------------
// Host launcher
// ---------------------------------------------------------------------------
extern "C" void kernel_launch(void* const* d_in, const int* in_sizes, int n_in,
                              void* d_out, int out_size, void* d_ws, size_t ws_size,
                              hipStream_t stream) {
    const float*     obj   = (const float*)d_in[0];
    const float*     pred  = (const float*)d_in[1];
    const long long* edges = (const long long*)d_in[2];
    const float* n1_w1 = (const float*)d_in[3];
    const float* n1_b1 = (const float*)d_in[4];
    const float* n1_w2 = (const float*)d_in[5];
    const float* n1_b2 = (const float*)d_in[6];
    const float* n2_w1 = (const float*)d_in[7];
    const float* n2_b1 = (const float*)d_in[8];
    const float* n2_w2 = (const float*)d_in[9];
    const float* n2_b2 = (const float*)d_in[10];

    float* out_obj = (float*)d_out;                                  // (B,N,DOUT)
    float* out_p   = (float*)d_out + (long long)kM2 * kDOUT;         // (B,T,DOUT)

    // workspace layout (bytes)
    char* ws = (char*)d_ws;
    unsigned short* curT   = (unsigned short*)(ws);                                  // M1 x K1
    unsigned short* w1t    = (unsigned short*)(ws + 100663296);                      // 512 x 1536
    unsigned short* w2t    = (unsigned short*)(ws + 100663296 + 1572864);            // 1536 x 512
    unsigned short* w3t    = (unsigned short*)(ws + 100663296 + 2 * 1572864);        // 512 x 512
    unsigned short* w4t    = (unsigned short*)(ws + 100663296 + 2 * 1572864 + 524288);
    unsigned short* hid    = (unsigned short*)(ws + 104857600);                      // M1 x H
    float*          pooled = (float*)(ws + 138412032);                               // M2 x H
    float*          counts = (float*)(ws + 171966464);                               // M2
    unsigned short* pbf    = (unsigned short*)(ws + 172032000);                      // M2 x H
    unsigned short* h2     = (unsigned short*)(ws + 188809216);                      // M2 x H

    // 1) weights -> bf16, transposed (N x K)
    k_w_cvt<<<(kK1 * kH + 255) / 256, 256, 0, stream>>>(n1_w1, w1t, kK1, kH);
    k_w_cvt<<<(kH * kN2 + 255) / 256, 256, 0, stream>>>(n1_w2, w2t, kH, kN2);
    k_w_cvt<<<(kH * kH  + 255) / 256, 256, 0, stream>>>(n2_w1, w3t, kH, kH);
    k_w_cvt<<<(kH * kDOUT + 255) / 256, 256, 0, stream>>>(n2_w2, w4t, kH, kDOUT);

    // 2) gather + convert edge activations
    k_gather<<<kM1, 256, 0, stream>>>(obj, pred, edges, curT);

    // 3) zero pooled + counts (contiguous), then counts
    long long nz = (long long)kM2 * kH + kM2;
    k_zero<<<(unsigned)((nz + 255) / 256), 256, 0, stream>>>(pooled, nz);
    k_counts<<<(kM1 + 255) / 256, 256, 0, stream>>>(edges, counts);

    // 4) GEMM1: hid = relu(curT @ n1_w1 + b1)   [32768 x 1536] x [1536 x 512]
    k_gemm<EPI_BF16><<<dim3(kM1 / 128, kH / 128), 256, 0, stream>>>(
        curT, w1t, n1_b1, kH, kK1, nullptr, hid, nullptr, nullptr, nullptr);

    // 5) GEMM2 + fused split/scatter: new_t = relu(hid @ n1_w2 + b2)
    k_gemm<EPI_SCATTER><<<dim3(kM1 / 128, kN2 / 128), 256, 0, stream>>>(
        hid, w2t, n1_b2, kN2, kH, nullptr, nullptr, edges, pooled, out_p);

    // 6) pooled / max(counts,1) -> bf16
    k_norm<<<(unsigned)(((long long)kM2 * kH + 255) / 256), 256, 0, stream>>>(
        pooled, counts, pbf);

    // 7) GEMM3: h2 = relu(pooled @ n2_w1 + b1)
    k_gemm<EPI_BF16><<<dim3(kM2 / 128, kH / 128), 256, 0, stream>>>(
        pbf, w3t, n2_b1, kH, kH, nullptr, h2, nullptr, nullptr, nullptr);

    // 8) GEMM4: new_obj = relu(h2 @ n2_w2 + b2) -> f32 output
    k_gemm<EPI_F32><<<dim3(kM2 / 128, kDOUT / 128), 256, 0, stream>>>(
        h2, w4t, n2_b2, kDOUT, kH, out_obj, nullptr, nullptr, nullptr, nullptr);
}
